// RNN_LSTM_764504178976
// MI455X (gfx1250) — matrix operand order repaired and checked by reading, not compiled, not run on hardware
//
#include <hip/hip_runtime.h>
#include <hip/hip_bf16.h>

typedef __attribute__((ext_vector_type(16))) __bf16 v16bf;
typedef __attribute__((ext_vector_type(8)))  __bf16 bf16x8;
typedef __attribute__((ext_vector_type(8)))  float  v8f;

namespace {
constexpr int kT   = 512;          // timesteps
constexpr int kB   = 64;           // batch
constexpr int kI   = 512;          // input size
constexpr int kH   = 1024;         // hidden
constexpr int kC   = 256;          // classes
constexpr int kKtot = kI + kH;     // fused K = 1536 ([x_t | h_{t-1}])
constexpr int kKP   = kKtot + 8;   // padded LDS row stride (bank-conflict avoidance)
constexpr int kNWG  = 64;          // persistent workgroups (16 hidden units each)
}

union AFrag { v16bf v; bf16x8 h[2]; };

__device__ __forceinline__ float fast_sigmoid(float x) {
  return 1.0f / (1.0f + __expf(-x));
}
__device__ __forceinline__ float fast_tanh(float x) {
  return 2.0f / (1.0f + __expf(-2.0f * x)) - 1.0f;
}

// ---------------------------------------------------------------------------
// Prep: x [B,T,I] f32 -> xT [T,B,I] bf16 (contiguous per-timestep A rows)
// ---------------------------------------------------------------------------
__global__ void prep_x_kernel(const float* __restrict__ x,
                              __hip_bfloat16* __restrict__ xT) {
  size_t i = (size_t)blockIdx.x * blockDim.x + threadIdx.x;
  size_t total = (size_t)kB * kT * kI;
  if (i >= total) return;
  int ii = (int)(i % kI);
  size_t bt = i / kI;
  int t = (int)(bt % kT);
  int b = (int)(bt / kT);
  xT[((size_t)t * kB + b) * kI + ii] = __float2bfloat16(x[i]);
}

// ---------------------------------------------------------------------------
// Prep: fuse W_ih|W_hh -> Wc [4H, 1536] bf16; W_fc -> bf16; h0 = 0; barrier = 0
// ---------------------------------------------------------------------------
__global__ void prep_w_kernel(const float* __restrict__ Wih,
                              const float* __restrict__ Whh,
                              const float* __restrict__ Wfc,
                              __hip_bfloat16* __restrict__ Wc,
                              __hip_bfloat16* __restrict__ Wfcb,
                              __hip_bfloat16* __restrict__ h0,
                              unsigned long long* __restrict__ bar) {
  size_t i = (size_t)blockIdx.x * blockDim.x + threadIdx.x;
  size_t nWc = (size_t)(4 * kH) * kKtot;
  if (i < nWc) {
    int r = (int)(i / kKtot);
    int k = (int)(i % kKtot);
    float v = (k < kI) ? Wih[(size_t)r * kI + k] : Whh[(size_t)r * kH + (k - kI)];
    Wc[i] = __float2bfloat16(v);
  }
  if (i < (size_t)kC * kH) Wfcb[i] = __float2bfloat16(Wfc[i]);
  if (i < (size_t)kB * kH) h0[i] = __float2bfloat16(0.0f);
  if (i == 0) *bar = 0ull;
}

// ---------------------------------------------------------------------------
// Persistent LSTM: 64 WGs x 128 threads (4 wave32). WG w owns hidden units
// [16w,16w+16): 64 gate rows (i,f,g,o x 16) of Wc held in LDS (bf16, padded).
// Per step: gates[64B x 64N] = [x_t | h] (K=1536) x Wc_chunk^T via
// v_wmma_f32_16x16x32_bf16; activations per-lane; h -> global (double buffer);
// device-wide atomic barrier between steps.
//
// LDS addressing: gate rows are (g*16 + n); gate +1 is 16*kKP elements
// (49408 B) away -> fits the 16-bit DS immediate. We keep TWO live base
// VGPRs (gates 0/1, gates 2/3); the second is pinned with an empty inline-asm
// so the compiler cannot fold it into base+0x18200 (which would exceed the
// immediate range and re-introduce per-iteration VALU address math + the
// WMMA->VALU hazard nops).
// ---------------------------------------------------------------------------
__launch_bounds__(128, 1)
__global__ void lstm_persistent_kernel(
    const __hip_bfloat16* __restrict__ xT,   // [T][B][I]
    const __hip_bfloat16* __restrict__ Wc,   // [4H][Ktot]
    const float* __restrict__ bih,
    const float* __restrict__ bhh,
    __hip_bfloat16* __restrict__ hbuf,       // [2][B][H]
    __hip_bfloat16* __restrict__ hs,         // [B][T][H]
    unsigned long long* __restrict__ bar) {
  extern __shared__ __hip_bfloat16 lw[];     // [64][kKP]
  const int wg   = blockIdx.x;
  const int tid  = threadIdx.x;
  const int wave = tid >> 5;
  const int lane = tid & 31;
  const int n    = lane & 15;
  const int hi   = lane >> 4;

  // Cooperative LDS fill: row rr in [0,64): gate = rr>>4, hidden = wg*16+(rr&15)
  for (int idx = tid; idx < 64 * (kKtot / 8); idx += 128) {
    int rr = idx / (kKtot / 8);
    int kk = (idx % (kKtot / 8)) * 8;
    int grow = (rr >> 4) * kH + wg * 16 + (rr & 15);
    *(bf16x8*)&lw[(size_t)rr * kKP + kk] =
        *(const bf16x8*)&Wc[(size_t)grow * kKtot + kk];
  }
  __syncthreads();

  const int j = wg * 16 + n;                 // hidden column for this lane's D
  float bias[4];
#pragma unroll
  for (int g = 0; g < 4; ++g) bias[g] = bih[g * kH + j] + bhh[g * kH + j];

  const int aRow  = wave * 16 + n;           // batch row fetched for A frags
  const int khalf = hi * 16;                 // A-layout K-half per lane group

  // Two loop-invariant LDS bases; gate +1 reached via 16*kKP immediate.
  const __hip_bfloat16* lwg01 = &lw[(size_t)(0 * 16 + n) * kKP + hi * 16];
  int base23 = (2 * 16 + n) * kKP + hi * 16;
  asm volatile("" : "+v"(base23));           // pin independent base VGPR
  const __hip_bfloat16* lwg23 = &lw[base23];

  float cst[8];                              // cell state: c[r+8*hi][j]
#pragma unroll
  for (int r = 0; r < 8; ++r) cst[r] = 0.0f;

  for (int t = 0; t < kT; ++t) {
    const __hip_bfloat16* hcur = hbuf + (size_t)(t & 1) * kB * kH;
    __hip_bfloat16* hnxt       = hbuf + (size_t)((t + 1) & 1) * kB * kH;

    v8f acc[4];
#pragma unroll
    for (int g = 0; g < 4; ++g)
#pragma unroll
      for (int r = 0; r < 8; ++r) acc[g][r] = 0.0f;

    const __hip_bfloat16* xrow = xT + ((size_t)t * kB + aRow) * kI;
    const __hip_bfloat16* hrow = hcur + (size_t)aRow * kH;

    // One fused-K step: load A + all 4 B fragments into distinct registers,
    // then issue 4 independent WMMAs (lets loads clause / overlap compute).
    auto kstep = [&](const __hip_bfloat16* ap, int kw) {
      AFrag a, b0, b1, b2, b3;
      a.h[0]  = *(const bf16x8*)(ap + khalf);
      a.h[1]  = *(const bf16x8*)(ap + khalf + 8);
      b0.h[0] = *(const bf16x8*)(lwg01 + kw);
      b0.h[1] = *(const bf16x8*)(lwg01 + kw + 8);
      b1.h[0] = *(const bf16x8*)(lwg01 + 16 * kKP + kw);
      b1.h[1] = *(const bf16x8*)(lwg01 + 16 * kKP + kw + 8);
      b2.h[0] = *(const bf16x8*)(lwg23 + kw);
      b2.h[1] = *(const bf16x8*)(lwg23 + kw + 8);
      b3.h[0] = *(const bf16x8*)(lwg23 + 16 * kKP + kw);
      b3.h[1] = *(const bf16x8*)(lwg23 + 16 * kKP + kw + 8);
      acc[0] = __builtin_amdgcn_wmma_f32_16x16x32_bf16(false, a.v, false, b0.v,
                                                       (short)0, acc[0], false, false);
      acc[1] = __builtin_amdgcn_wmma_f32_16x16x32_bf16(false, a.v, false, b1.v,
                                                       (short)0, acc[1], false, false);
      acc[2] = __builtin_amdgcn_wmma_f32_16x16x32_bf16(false, a.v, false, b2.v,
                                                       (short)0, acc[2], false, false);
      acc[3] = __builtin_amdgcn_wmma_f32_16x16x32_bf16(false, a.v, false, b3.v,
                                                       (short)0, acc[3], false, false);
    };

    // x-phase: K = 0..511 (A rows from xT), fully unrolled -> immediate offsets.
#pragma unroll
    for (int k0 = 0; k0 < kI; k0 += 32) kstep(xrow + k0, k0);
    // h-phase: K = 512..1535 (A rows from h double buffer).
#pragma unroll
    for (int k0 = 0; k0 < kH; k0 += 32) kstep(hrow + k0, kI + k0);

    // Activations: i,f,g,o coincide per (lane, r) across the 4 accumulators.
#pragma unroll
    for (int r = 0; r < 8; ++r) {
      float iv = acc[0][r] + bias[0];
      float fv = acc[1][r] + bias[1];
      float gv = acc[2][r] + bias[2];
      float ov = acc[3][r] + bias[3];
      float cn = fast_sigmoid(fv) * cst[r] + fast_sigmoid(iv) * fast_tanh(gv);
      cst[r] = cn;
      float hn = fast_sigmoid(ov) * fast_tanh(cn);
      int mD = wave * 16 + r + 8 * hi;       // batch row in D layout
      __hip_bfloat16 hb = __float2bfloat16(hn);
      hnxt[(size_t)mD * kH + j] = hb;
      hs[((size_t)mD * kT + t) * kH + j] = hb;
    }

    // Device-wide barrier: release h writes, arrive, spin (s_sleep), resync.
    __threadfence();
    __syncthreads();
    if (tid == 0) {
      __hip_atomic_fetch_add(bar, 1ull, __ATOMIC_ACQ_REL, __HIP_MEMORY_SCOPE_AGENT);
      unsigned long long target = (unsigned long long)(t + 1) * kNWG;
      while (__hip_atomic_load(bar, __ATOMIC_ACQUIRE, __HIP_MEMORY_SCOPE_AGENT) < target)
        __builtin_amdgcn_s_sleep(1);
    }
    __syncthreads();
  }
}

// ---------------------------------------------------------------------------
// FC: out[m][c] = hs[m][:] . Wfc[c][:] + b_fc[c]; M=32768, N=256, K=1024.
// Grid (M/64, N/64), 4 waves/WG, each wave a 16x64 strip (4 WMMA tiles).
// All 4 N-tiles of W_fc addressed from ONE base via 24-bit global immediates.
// ---------------------------------------------------------------------------
__launch_bounds__(128, 4)
__global__ void fc_kernel(const __hip_bfloat16* __restrict__ hs,    // [32768][1024]
                          const __hip_bfloat16* __restrict__ Wfcb,  // [256][1024]
                          const float* __restrict__ bfc,
                          float* __restrict__ out) {                // [32768][256]
  const int tid = threadIdx.x, wave = tid >> 5, lane = tid & 31;
  const int n = lane & 15, hi = lane >> 4;
  const int m0 = blockIdx.x * 64 + wave * 16;
  const int c0 = blockIdx.y * 64;
  const int aRow  = m0 + n;
  const int khalf = hi * 16;

  v8f acc[4];
#pragma unroll
  for (int g = 0; g < 4; ++g)
#pragma unroll
    for (int r = 0; r < 8; ++r) acc[g][r] = 0.0f;

  const __hip_bfloat16* arow  = hs + (size_t)aRow * kH;
  const __hip_bfloat16* wbase = Wfcb + (size_t)(c0 + n) * kH + hi * 16;

#pragma unroll 4
  for (int k0 = 0; k0 < kH; k0 += 32) {
    AFrag a, b0, b1, b2, b3;
    a.h[0]  = *(const bf16x8*)(arow + k0 + khalf);
    a.h[1]  = *(const bf16x8*)(arow + k0 + khalf + 8);
    b0.h[0] = *(const bf16x8*)(wbase + 0 * 16 * kH + k0);
    b0.h[1] = *(const bf16x8*)(wbase + 0 * 16 * kH + k0 + 8);
    b1.h[0] = *(const bf16x8*)(wbase + 1 * 16 * kH + k0);
    b1.h[1] = *(const bf16x8*)(wbase + 1 * 16 * kH + k0 + 8);
    b2.h[0] = *(const bf16x8*)(wbase + 2 * 16 * kH + k0);
    b2.h[1] = *(const bf16x8*)(wbase + 2 * 16 * kH + k0 + 8);
    b3.h[0] = *(const bf16x8*)(wbase + 3 * 16 * kH + k0);
    b3.h[1] = *(const bf16x8*)(wbase + 3 * 16 * kH + k0 + 8);
    acc[0] = __builtin_amdgcn_wmma_f32_16x16x32_bf16(false, a.v, false, b0.v,
                                                     (short)0, acc[0], false, false);
    acc[1] = __builtin_amdgcn_wmma_f32_16x16x32_bf16(false, a.v, false, b1.v,
                                                     (short)0, acc[1], false, false);
    acc[2] = __builtin_amdgcn_wmma_f32_16x16x32_bf16(false, a.v, false, b2.v,
                                                     (short)0, acc[2], false, false);
    acc[3] = __builtin_amdgcn_wmma_f32_16x16x32_bf16(false, a.v, false, b3.v,
                                                     (short)0, acc[3], false, false);
  }
#pragma unroll
  for (int nt = 0; nt < 4; ++nt) {
    int cc = c0 + nt * 16 + n;
    float bv = bfc[cc];
#pragma unroll
    for (int r = 0; r < 8; ++r) {
      int mD = m0 + r + 8 * hi;
      out[(size_t)mD * kC + cc] = acc[nt][r] + bv;
    }
  }
}

// ---------------------------------------------------------------------------
extern "C" void kernel_launch(void* const* d_in, const int* in_sizes, int n_in,
                              void* d_out, int out_size, void* d_ws, size_t ws_size,
                              hipStream_t stream) {
  const float* x   = (const float*)d_in[0];
  const float* Wih = (const float*)d_in[1];
  const float* Whh = (const float*)d_in[2];
  const float* bih = (const float*)d_in[3];
  const float* bhh = (const float*)d_in[4];
  const float* Wfc = (const float*)d_in[5];
  const float* bfc = (const float*)d_in[6];
  float* out = (float*)d_out;

  // Workspace carve-out (256B aligned): ~109 MB total.
  char* p = (char*)d_ws;
  auto carve = [&](size_t bytes) {
    char* q = p;
    p += (bytes + 255) & ~(size_t)255;
    return q;
  };
  __hip_bfloat16* xT   = (__hip_bfloat16*)carve((size_t)kT * kB * kI * 2);      // 32 MB
  __hip_bfloat16* Wc   = (__hip_bfloat16*)carve((size_t)4 * kH * kKtot * 2);    // 12 MB
  __hip_bfloat16* Wfcb = (__hip_bfloat16*)carve((size_t)kC * kH * 2);           // 0.5 MB
  __hip_bfloat16* hbuf = (__hip_bfloat16*)carve((size_t)2 * kB * kH * 2);       // 256 KB
  unsigned long long* bar = (unsigned long long*)carve(256);
  __hip_bfloat16* hs   = (__hip_bfloat16*)carve((size_t)kB * kT * kH * 2);      // 64 MB
  (void)ws_size; (void)in_sizes; (void)n_in; (void)out_size;

  // 1) Convert/transpose x.
  {
    size_t total = (size_t)kB * kT * kI;
    int blocks = (int)((total + 255) / 256);
    prep_x_kernel<<<blocks, 256, 0, stream>>>(x, xT);
  }
  // 2) Fuse/convert weights, zero h0 + barrier.
  {
    size_t total = (size_t)4 * kH * kKtot;
    int blocks = (int)((total + 255) / 256);
    prep_w_kernel<<<blocks, 256, 0, stream>>>(Wih, Whh, Wfc, Wc, Wfcb, hbuf, bar);
  }
  // 3) Persistent LSTM recurrence: 64 WGs, 193 KB LDS each (CDNA5: 320 KB/WGP).
  {
    size_t ldsBytes = (size_t)64 * kKP * sizeof(__hip_bfloat16);  // 197632
    (void)hipFuncSetAttribute((const void*)lstm_persistent_kernel,
                              hipFuncAttributeMaxDynamicSharedMemorySize,
                              (int)ldsBytes);
    lstm_persistent_kernel<<<kNWG, 128, ldsBytes, stream>>>(xT, Wc, bih, bhh,
                                                            hbuf, hs, bar);
  }
  // 4) Final FC GEMM.
  {
    dim3 grid((kB * kT) / 64, kC / 64);
    fc_kernel<<<grid, 128, 0, stream>>>(hs, Wfcb, bfc, out);
  }
}
